// attention_mod_31396210934284
// MI455X (gfx1250) — matrix operand order repaired and checked by reading, not compiled, not run on hardware
//
#include <hip/hip_runtime.h>
#include <math.h>

typedef __bf16 bf16;
typedef __attribute__((ext_vector_type(16))) bf16  v16bf;
typedef __attribute__((ext_vector_type(8)))  bf16  v8bf;
typedef __attribute__((ext_vector_type(8)))  float v8f;

#define HDIM 1024
#define TDIM 2048
#define BDIM 32
#define MB   64            // rows of (b,t) per workgroup
#define LDA  1032          // padded bf16 LDS stride (+16B -> 4-bank rotation/row)

union ABFrag { v16bf v; v8bf h[2]; };

// Fast tanh: native V_TANH_F32 on gfx1250 if clang exposes it, else 2-trans-op
// formulation tanh(x) = 1 - 2/(exp(2x)+1) using v_exp_f32 + v_rcp_f32.
__device__ __forceinline__ float fast_tanh(float x) {
#if __has_builtin(__builtin_amdgcn_tanhf)
  return __builtin_amdgcn_tanhf(x);
#else
  const float e = __expf(2.0f * x);
  return 1.0f - 2.0f * __builtin_amdgcn_rcpf(e + 1.0f);
#endif
}

// ---------------------------------------------------------------------------
// Kernel 1: W2 (second half of W columns) fp32 -> bf16, row-major [H][H]
// ---------------------------------------------------------------------------
__global__ __launch_bounds__(256)
void wconv_kernel(const float* __restrict__ W, bf16* __restrict__ Wb) {
  const int idx = (blockIdx.x * 256 + threadIdx.x) * 4;   // over H*H elements
  const int n = idx >> 10;
  const int k = idx & (HDIM - 1);
  const float4 f = *(const float4*)(W + (size_t)n * (2 * HDIM) + HDIM + k);
  union { bf16 b[4]; uint2 u; } pk;
  pk.b[0] = (bf16)f.x; pk.b[1] = (bf16)f.y; pk.b[2] = (bf16)f.z; pk.b[3] = (bf16)f.w;
  *(uint2*)(Wb + idx) = pk.u;
}

// ---------------------------------------------------------------------------
// Kernel 2: C[b,h] = bias[h] + hidden[b,:] . W[h, 0:H]   (t-invariant part)
// ---------------------------------------------------------------------------
__global__ __launch_bounds__(256)
void pre1_kernel(const float* __restrict__ hidden, const float* __restrict__ W,
                 const float* __restrict__ bias, float* __restrict__ C) {
  const int idx = blockIdx.x * 256 + threadIdx.x;   // B*H threads
  const int h = idx & (HDIM - 1);
  const int b = idx >> 10;
  const float* hp = hidden + (size_t)b * HDIM;
  const float* wp = W + (size_t)h * (2 * HDIM);
  float acc = bias[h];
  for (int d = 0; d < HDIM; d += 4) {
    const float4 x = *(const float4*)(hp + d);
    const float4 w = *(const float4*)(wp + d);
    acc += x.x * w.x + x.y * w.y + x.z * w.z + x.w * w.w;
  }
  C[idx] = acc;
}

// ---------------------------------------------------------------------------
// Kernel 3: fused GEMM (bf16 WMMA, fp32 acc) + tanh + v-dot -> scores[B*T]
// ---------------------------------------------------------------------------
__global__ __launch_bounds__(256)
void attn_gemm_kernel(const float* __restrict__ enc, const bf16* __restrict__ Wb,
                      const float* __restrict__ C, const float* __restrict__ Vv,
                      float* __restrict__ scores) {
  __shared__ bf16  As[MB * LDA];
  __shared__ float s_score[MB];

  const int tid = threadIdx.x;
  const int row_base = blockIdx.x * MB;     // MB divides TDIM -> single batch b
  const int bb = row_base >> 11;            // row_base / TDIM

  if (tid < MB) s_score[tid] = 0.0f;

  // Stage A block: 64 rows x 1024 cols, fp32 -> bf16 into padded LDS
  const float* encBase = enc + (size_t)row_base * HDIM;
  for (int e = tid * 4; e < MB * HDIM; e += 256 * 4) {
    const int row = e >> 10;
    const int col = e & (HDIM - 1);
    const float4 f = *(const float4*)(encBase + (size_t)row * HDIM + col);
    union { bf16 b[4]; uint2 u; } pk;
    pk.b[0] = (bf16)f.x; pk.b[1] = (bf16)f.y; pk.b[2] = (bf16)f.z; pk.b[3] = (bf16)f.w;
    *(uint2*)(&As[row * LDA + col]) = pk.u;
  }
  __syncthreads();

  const int lane = tid & 31;
  const int wid  = tid >> 5;
  const int msub = wid & 3;                 // 4 M sub-tiles of 16 rows
  const int nh   = wid >> 2;                // 2 N halves of 512 cols
  const int l16  = lane & 15;
  const int khA  = (lane >> 4) * 8;         // A frag K-interleave (ISA layout)
  const int khB  = (lane >> 4) * 8;         // B frag K-interleave (ISA layout)
  const bf16* aRow = &As[(msub * 16 + l16) * LDA];

  float p[8];
  #pragma unroll
  for (int r = 0; r < 8; ++r) p[r] = 0.0f;

  for (int g = 0; g < 4; ++g) {             // N groups: 8 tiles (128 cols) each
    const int ntb0 = nh * 512 + g * 128;
    v8f acc[8] = {};

    for (int k0 = 0; k0 < HDIM; k0 += 32) {
      ABFrag a;
      a.h[0] = *(const v8bf*)(aRow + k0 + khA);
      a.h[1] = *(const v8bf*)(aRow + k0 + 16 + khA);
      #pragma unroll
      for (int j = 0; j < 8; ++j) {
        const bf16* bp = Wb + (size_t)(ntb0 + j * 16 + l16) * HDIM + k0 + khB;
        ABFrag b;
        b.h[0] = *(const v8bf*)bp;
        b.h[1] = *(const v8bf*)(bp + 16);
        acc[j] = __builtin_amdgcn_wmma_f32_16x16x32_bf16(
            false, a.v, false, b.v, (short)0, acc[j], false, false);
      }
    }

    // Fused epilogue: p[r] += v[n] * tanh(acc + C[b,n])
    #pragma unroll
    for (int j = 0; j < 8; ++j) {
      const int n = ntb0 + j * 16 + l16;
      const float cb = C[bb * HDIM + n];
      const float vv = Vv[n];
      #pragma unroll
      for (int r = 0; r < 8; ++r)
        p[r] = fmaf(vv, fast_tanh(acc[j][r] + cb), p[r]);
    }
  }

  // Reduce over N (16 lanes per half; halves hold rows 0-7 / 8-15 of sub-tile)
  #pragma unroll
  for (int r = 0; r < 8; ++r) {
    #pragma unroll
    for (int m = 1; m < 16; m <<= 1)
      p[r] += __shfl_xor(p[r], m, 32);
  }
  if (l16 == 0) {
    const int rb = msub * 16 + ((lane >= 16) ? 8 : 0);
    #pragma unroll
    for (int r = 0; r < 8; ++r) atomicAdd(&s_score[rb + r], p[r]);
  }
  __syncthreads();
  if (tid < MB) scores[row_base + tid] = s_score[tid];
}

// ---------------------------------------------------------------------------
// Kernel 4: softmax over T per batch row
// ---------------------------------------------------------------------------
__global__ __launch_bounds__(256)
void softmax_kernel(const float* __restrict__ scores, float* __restrict__ out) {
  __shared__ float red[256];
  const int b = blockIdx.x;
  const int tid = threadIdx.x;
  const float* s = scores + (size_t)b * TDIM;

  float vals[8];
  float mx = -INFINITY;
  #pragma unroll
  for (int i = 0; i < 8; ++i) { vals[i] = s[tid + i * 256]; mx = fmaxf(mx, vals[i]); }
  red[tid] = mx; __syncthreads();
  for (int off = 128; off > 0; off >>= 1) {
    if (tid < off) red[tid] = fmaxf(red[tid], red[tid + off]);
    __syncthreads();
  }
  mx = red[0]; __syncthreads();

  float sum = 0.0f;
  #pragma unroll
  for (int i = 0; i < 8; ++i) { vals[i] = __expf(vals[i] - mx); sum += vals[i]; }
  red[tid] = sum; __syncthreads();
  for (int off = 128; off > 0; off >>= 1) {
    if (tid < off) red[tid] += red[tid + off];
    __syncthreads();
  }
  const float inv = __builtin_amdgcn_rcpf(red[0]);
  #pragma unroll
  for (int i = 0; i < 8; ++i) out[(size_t)b * TDIM + tid + i * 256] = vals[i] * inv;
}

// ---------------------------------------------------------------------------
extern "C" void kernel_launch(void* const* d_in, const int* in_sizes, int n_in,
                              void* d_out, int out_size, void* d_ws, size_t ws_size,
                              hipStream_t stream) {
  (void)in_sizes; (void)n_in; (void)out_size; (void)ws_size;
  const float* hidden = (const float*)d_in[0];   // [1,B,H]
  const float* enc    = (const float*)d_in[1];   // [B,T,H]
  const float* W      = (const float*)d_in[2];   // [H,2H]
  const float* bias   = (const float*)d_in[3];   // [H]
  const float* v      = (const float*)d_in[4];   // [H]
  float* out = (float*)d_out;                    // [B,1,T]

  char* ws = (char*)d_ws;
  bf16*  Wb = (bf16*)ws;                                   // 2 MB bf16 W2
  float* C  = (float*)(ws + (size_t)2 * 1024 * 1024);      // 128 KB  [B,H]
  float* sc = (float*)(ws + (size_t)2 * 1024 * 1024 + 128 * 1024); // 256 KB [B,T]

  wconv_kernel   <<<(HDIM * HDIM) / (4 * 256), 256, 0, stream>>>(W, Wb);
  pre1_kernel    <<<(BDIM * HDIM) / 256,       256, 0, stream>>>(hidden, W, bias, C);
  attn_gemm_kernel<<<(BDIM * TDIM) / MB,       256, 0, stream>>>(enc, Wb, C, v, sc);
  softmax_kernel <<<BDIM,                      256, 0, stream>>>(sc, out);
}
